// GraphTransformerEncoderLayer_80668075753649
// MI455X (gfx1250) — matrix operand order repaired and checked by reading, not compiled, not run on hardware
//
#include <hip/hip_runtime.h>

typedef __attribute__((ext_vector_type(16))) _Float16 v16h;
typedef __attribute__((ext_vector_type(8)))  _Float16 v8h;
typedef __attribute__((ext_vector_type(2)))  _Float16 h2;
typedef __attribute__((ext_vector_type(8)))  float    v8f;

#define DEVINL __device__ __forceinline__

DEVINL void fadd_atomic(float* p, float v) {
    // relaxed + agent scope -> hardware global_atomic_add_f32 (resolves in L2)
    __hip_atomic_fetch_add(p, v, __ATOMIC_RELAXED, __HIP_MEMORY_SCOPE_AGENT);
}

// ---------------------------------------------------------------- utilities
__global__ void k_zero(float* __restrict__ p, long n) {
    long i = blockIdx.x * (long)blockDim.x + threadIdx.x;
    long s = (long)gridDim.x * blockDim.x;
    for (; i < n; i += s) p[i] = 0.0f;
}

__global__ void k_f32_to_f16(const float* __restrict__ in, _Float16* __restrict__ out, long n) {
    long i = blockIdx.x * (long)blockDim.x + threadIdx.x;
    long s = (long)gridDim.x * blockDim.x;
    for (; i < n; i += s) out[i] = (_Float16)in[i];
}

// ---------------------------------------------------------------- WMMA GEMM
// OUT[M,Nc] = A[M,K](f16) @ W[K,Nc](f16)  (+bias) (relu) (+resid)
// Epilogue configuration is compile-time so the inner loop is branch-free.
// M must be a multiple of 16 (50000 and 800000 both are).
template <int K, int Nc, bool BIAS, bool RESID, bool O32, bool O16, bool STATS, bool RELU>
__global__ __launch_bounds__(256)
void k_gemm(const _Float16* __restrict__ A, const _Float16* __restrict__ W,
            const float* __restrict__ bias, const float* __restrict__ resid,
            float* __restrict__ out32, _Float16* __restrict__ out16,
            float* __restrict__ ssum, float* __restrict__ ssq,
            int M)
{
    constexpr int KS = K / 32;                // WMMA k-steps
    constexpr int NT = Nc / 16;               // 16-col output tiles
    __shared__ _Float16 Wt[Nc * K];           // weights transposed: Wt[n*K + k]

    const int tid = threadIdx.x;
    for (int i = tid; i < K * Nc; i += 256) {
        int k = i / Nc, n = i - k * Nc;
        Wt[n * K + k] = W[i];
    }
    __syncthreads();

    const int lane = tid & 31;
    const int wave = tid >> 5;
    const int m    = lane & 15;               // A-row within tile / C-col within tile
    const int hs   = lane >> 4;               // lane-half select

    // ---- hoist ALL B fragments into registers (wave-invariant) ----
    v16h bf[KS][NT];
    float bc[NT];
#pragma unroll
    for (int nt = 0; nt < NT; ++nt) {
        const _Float16* Wn = &Wt[(nt * 16 + m) * K];
#pragma unroll
        for (int ks = 0; ks < KS; ++ks) {
#pragma unroll
            for (int j = 0; j < 8; ++j) {
                h2 p = *(const h2*)(Wn + ks * 32 + hs * 16 + 2 * j);
                bf[ks][nt][2 * j]     = p[0];
                bf[ks][nt][2 * j + 1] = p[1];
            }
        }
        if (BIAS) bc[nt] = bias[nt * 16 + m];
    }

    const int mtiles = M >> 4;
    const int gw     = blockIdx.x * 8 + wave;
    const int nw     = gridDim.x * 8;

    for (int rt = gw; rt < mtiles; rt += nw) {
        const _Float16* Arow = A + (long)(rt * 16 + m) * K;

        // ---- A fragments: loaded once per 16-row slab ----
        v16h af[KS];
#pragma unroll
        for (int ks = 0; ks < KS; ++ks) {
            const int ka = ks * 32 + hs * 16;
            v8h lo = *(const v8h*)(Arow + ka);
            v8h hi = *(const v8h*)(Arow + ka + 8);
            af[ks] = __builtin_shufflevector(lo, hi,
                     0,1,2,3,4,5,6,7,8,9,10,11,12,13,14,15);
        }

#pragma unroll
        for (int nt = 0; nt < NT; ++nt) {
            v8f acc;
#pragma unroll
            for (int i = 0; i < 8; ++i) acc[i] = 0.0f;
#pragma unroll
            for (int ks = 0; ks < KS; ++ks)
                acc = __builtin_amdgcn_wmma_f32_16x16x32_f16(
                        false, af[ks], false, bf[ks][nt], (short)0, acc, false, false);

            // epilogue: lane holds column ncol, rows i + 8*hs of the tile
            const int ncol = nt * 16 + m;
            const long o0  = (long)(rt * 16 + 8 * hs) * Nc + ncol;
            float lsum = 0.0f, lsq = 0.0f;
#pragma unroll
            for (int i = 0; i < 8; ++i) {
                const long o = o0 + (long)i * Nc;
                float v = acc[i];
                if (BIAS)  v += bc[nt];
                if (RELU)  v = v > 0.0f ? v : 0.0f;
                if (RESID) v += resid[o];
                if (O32)   out32[o] = v;
                if (O16)   out16[o] = (_Float16)v;
                if (STATS) { lsum += v; lsq += v * v; }
            }
            if (STATS) {
                // lanes L and L+16 hold the same column: combine, halve atomics
                lsum += __shfl_xor(lsum, 16, 32);
                lsq  += __shfl_xor(lsq, 16, 32);
                if (hs == 0) {
                    fadd_atomic(&ssum[ncol], lsum);
                    fadd_atomic(&ssq[ncol],  lsq);
                }
            }
        }
    }
}

// ---------------------------------------------------------------- per-edge attention
// one thread per (edge, head): score = q[dst]*k[src]*scale*e ; a = exp(clip(sum))
// scatter a*v[src] -> wv, a -> z with hw fp atomics (wv/z live in 192MB L2)
__global__ __launch_bounds__(256)
void k_edge(const _Float16* __restrict__ qh, const _Float16* __restrict__ kh,
            const _Float16* __restrict__ vh, const _Float16* __restrict__ eh,
            const int* __restrict__ ei, _Float16* __restrict__ score,
            float* __restrict__ wv, float* __restrict__ z, long EG)
{
    const float scale = 0.25f;                 // 1/sqrt(DH), DH=16
    long i = blockIdx.x * (long)blockDim.x + threadIdx.x;
    long s = (long)gridDim.x * blockDim.x;
    const long total = EG * 4;
    for (; i < total; i += s) {
        const long g = i >> 2;
        const int  h = (int)(i & 3);
        const int  src = ei[g];
        const int  dst = ei[EG + g];
        const v8h* qp = (const v8h*)(qh + (long)dst * 64 + h * 16);
        const v8h* kp = (const v8h*)(kh + (long)src * 64 + h * 16);
        const v8h* ep = (const v8h*)(eh + g * 64 + h * 16);
        v8h q0 = qp[0], q1 = qp[1];
        v8h k0 = kp[0], k1 = kp[1];
        v8h e0 = ep[0], e1 = ep[1];
        float sv[16];
        float tot = 0.0f;
#pragma unroll
        for (int j = 0; j < 8; ++j) {
            sv[j]     = (float)q0[j] * (float)k0[j] * scale * (float)e0[j];
            sv[8 + j] = (float)q1[j] * (float)k1[j] * scale * (float)e1[j];
            tot += sv[j] + sv[8 + j];
        }
        v8h s0, s1;
#pragma unroll
        for (int j = 0; j < 8; ++j) { s0[j] = (_Float16)sv[j]; s1[j] = (_Float16)sv[8 + j]; }
        v8h* sp = (v8h*)(score + g * 64 + h * 16);
        sp[0] = s0; sp[1] = s1;

        float cl = fminf(5.0f, fmaxf(-5.0f, tot));
        float a  = __expf(cl);

        const v8h* vp = (const v8h*)(vh + (long)src * 64 + h * 16);
        v8h v0 = vp[0], v1 = vp[1];
        float* wvp = wv + (long)dst * 64 + h * 16;
#pragma unroll
        for (int j = 0; j < 8; ++j) {
            fadd_atomic(wvp + j,     a * (float)v0[j]);
            fadd_atomic(wvp + 8 + j, a * (float)v1[j]);
        }
        fadd_atomic(z + (long)dst * 4 + h, a);
    }
}

// wv / (z + 1e-6) -> f16 for the @Oh GEMM
__global__ void k_nodenorm(const float* __restrict__ wv, const float* __restrict__ z,
                           _Float16* __restrict__ nh, long n) {
    long i = blockIdx.x * (long)blockDim.x + threadIdx.x;
    long s = (long)gridDim.x * blockDim.x;
    for (; i < n; i += s) {
        long node = i >> 6;
        int  h    = (int)((i >> 4) & 3);
        nh[i] = (_Float16)(wv[i] / (z[node * 4 + h] + 1e-6f));
    }
}

// turn (sum,sumsq) into (mean, rsqrt(var+eps))
__global__ void k_bnfin(float* __restrict__ ssum, float* __restrict__ ssq, float invN) {
    int c = threadIdx.x;
    if (c < 64) {
        float mean = ssum[c] * invN;
        float var  = ssq[c] * invN - mean * mean;
        ssum[c] = mean;
        ssq[c]  = __frsqrt_rn(var + 1e-5f);
    }
}

// y = (t-mean)*inv*g + b ; in-place f32, optional f16 copy for next GEMM
template <bool O16>
__global__ void k_bnapply(float* __restrict__ t, const float* __restrict__ mean,
                          const float* __restrict__ inv, const float* __restrict__ g,
                          const float* __restrict__ b, _Float16* __restrict__ out16, long n) {
    long i = blockIdx.x * (long)blockDim.x + threadIdx.x;
    long s = (long)gridDim.x * blockDim.x;
    for (; i < n; i += s) {
        int c = (int)(i & 63);
        float v = (t[i] - mean[c]) * inv[c] * g[c] + b[c];
        t[i] = v;
        if (O16) out16[i] = (_Float16)v;
    }
}

// ---------------------------------------------------------------- launcher
extern "C" void kernel_launch(void* const* d_in, const int* in_sizes, int n_in,
                              void* d_out, int out_size, void* d_ws, size_t ws_size,
                              hipStream_t stream) {
    (void)in_sizes; (void)n_in; (void)out_size; (void)ws_size;
    const long N = 50000, EG = 800000, D = 64, DF = 128;

    const float* x    = (const float*)d_in[0];
    const float* ea   = (const float*)d_in[1];
    const int*   ei   = (const int*)  d_in[2];
    const float* Wq   = (const float*)d_in[3];
    const float* Wk   = (const float*)d_in[4];
    const float* Wv   = (const float*)d_in[5];
    const float* We   = (const float*)d_in[6];
    const float* Oh   = (const float*)d_in[7];
    const float* Oe   = (const float*)d_in[8];
    const float* nx_g = (const float*)d_in[9];
    const float* nx_b = (const float*)d_in[10];
    const float* ne_g = (const float*)d_in[11];
    const float* ne_b = (const float*)d_in[12];
    const float* fnW1 = (const float*)d_in[13];
    const float* fnb1 = (const float*)d_in[14];
    const float* fnW2 = (const float*)d_in[15];
    const float* fnb2 = (const float*)d_in[16];
    const float* fn_g = (const float*)d_in[17];
    const float* fn_b = (const float*)d_in[18];
    const float* feW1 = (const float*)d_in[19];
    const float* feb1 = (const float*)d_in[20];
    const float* feW2 = (const float*)d_in[21];
    const float* feb2 = (const float*)d_in[22];
    const float* fe_g = (const float*)d_in[23];
    const float* fe_b = (const float*)d_in[24];

    char* base = (char*)d_ws;
    size_t off = 0;
    auto alloc = [&](size_t bytes) -> void* {
        void* p = base + off;
        off += (bytes + 255) & ~(size_t)255;
        return p;
    };

    _Float16* WqH   = (_Float16*)alloc(D * D * 2);
    _Float16* WkH   = (_Float16*)alloc(D * D * 2);
    _Float16* WvH   = (_Float16*)alloc(D * D * 2);
    _Float16* WeH   = (_Float16*)alloc(D * D * 2);
    _Float16* OhH   = (_Float16*)alloc(D * D * 2);
    _Float16* OeH   = (_Float16*)alloc(D * D * 2);
    _Float16* fnW1H = (_Float16*)alloc(D * DF * 2);
    _Float16* fnW2H = (_Float16*)alloc(DF * D * 2);
    _Float16* feW1H = (_Float16*)alloc(D * DF * 2);
    _Float16* feW2H = (_Float16*)alloc(DF * D * 2);
    _Float16* xh    = (_Float16*)alloc(N * D * 2);
    _Float16* qh    = (_Float16*)alloc(N * D * 2);
    _Float16* kh    = (_Float16*)alloc(N * D * 2);
    _Float16* vh    = (_Float16*)alloc(N * D * 2);
    _Float16* ehin  = (_Float16*)alloc(EG * D * 2);   // edge_attr f16 -> score f16 -> edge_n f16
    _Float16* eW    = (_Float16*)alloc(EG * D * 2);   // e = edge_attr @ We (f16)
    float*    wv    = (float*)   alloc(N * D * 4);
    float*    z     = (float*)   alloc(N * 4 * 4);
    _Float16* nh    = (_Float16*)alloc(N * D * 2);
    float*    nbuf  = (float*)   alloc(N * D * 4);    // node_out -> node_n (in place)
    _Float16* nn16  = (_Float16*)alloc(N * D * 2);
    _Float16* hidn  = (_Float16*)alloc(N * DF * 2);
    float*    ebuf  = (float*)   alloc(EG * D * 4);   // edge_out -> edge_n (in place)
    _Float16* hide  = (_Float16*)alloc(EG * DF * 2);
    float*    stats = (float*)   alloc(512 * 4);      // 4 pairs of (sum[64], sumsq[64])

    float* nodeOut = (float*)d_out;
    float* edgeOut = (float*)d_out + N * D;

    auto grid1 = [](long n) { long b = (n + 255) / 256; return (int)(b > 8192 ? 8192 : b); };
    auto gemmGrid = [](long M) {
        int blocks = (int)((M / 16 + 7) / 8);
        if (blocks > 4096) blocks = 4096;
        if (blocks < 1) blocks = 1;
        return blocks;
    };

    // zero accumulators (workspace is poisoned, must re-zero every call)
    k_zero<<<grid1(N * D), 256, 0, stream>>>(wv, N * D);
    k_zero<<<grid1(N * 4), 256, 0, stream>>>(z, N * 4);
    k_zero<<<1, 256, 0, stream>>>(stats, 512);

    // f16 staging
    k_f32_to_f16<<<grid1(N * D), 256, 0, stream>>>(x, xh, N * D);
    k_f32_to_f16<<<grid1(EG * D), 256, 0, stream>>>(ea, ehin, EG * D);
    k_f32_to_f16<<<16, 256, 0, stream>>>(Wq, WqH, D * D);
    k_f32_to_f16<<<16, 256, 0, stream>>>(Wk, WkH, D * D);
    k_f32_to_f16<<<16, 256, 0, stream>>>(Wv, WvH, D * D);
    k_f32_to_f16<<<16, 256, 0, stream>>>(We, WeH, D * D);
    k_f32_to_f16<<<16, 256, 0, stream>>>(Oh, OhH, D * D);
    k_f32_to_f16<<<16, 256, 0, stream>>>(Oe, OeH, D * D);
    k_f32_to_f16<<<32, 256, 0, stream>>>(fnW1, fnW1H, D * DF);
    k_f32_to_f16<<<32, 256, 0, stream>>>(fnW2, fnW2H, DF * D);
    k_f32_to_f16<<<32, 256, 0, stream>>>(feW1, feW1H, D * DF);
    k_f32_to_f16<<<32, 256, 0, stream>>>(feW2, feW2H, DF * D);

    // ---- GEMM configurations (all compile-time) ----
    //                K   Nc  BIAS  RESID O32   O16   STATS RELU
    // proj:          64  64  no    no    no    yes   no    no
    // attn-out:      64  64  no    yes   yes   no    yes   no
    // ffn-fc1:       64  128 yes   no    no    yes   no    yes
    // ffn-fc2:       128 64  yes   yes   yes   no    yes   no

    // q/k/v/e projections (f16 out for cheap gathers)
    k_gemm<64, 64, false, false, false, true, false, false>
        <<<gemmGrid(N), 256, 0, stream>>>(xh, WqH, nullptr, nullptr,
            nullptr, qh, nullptr, nullptr, (int)N);
    k_gemm<64, 64, false, false, false, true, false, false>
        <<<gemmGrid(N), 256, 0, stream>>>(xh, WkH, nullptr, nullptr,
            nullptr, kh, nullptr, nullptr, (int)N);
    k_gemm<64, 64, false, false, false, true, false, false>
        <<<gemmGrid(N), 256, 0, stream>>>(xh, WvH, nullptr, nullptr,
            nullptr, vh, nullptr, nullptr, (int)N);
    k_gemm<64, 64, false, false, false, true, false, false>
        <<<gemmGrid(EG), 256, 0, stream>>>(ehin, WeH, nullptr, nullptr,
            nullptr, eW, nullptr, nullptr, (int)EG);

    // per-edge scores + scatter (score overwrites ehin)
    k_edge<<<8192, 256, 0, stream>>>(qh, kh, vh, eW, ei, ehin, wv, z, EG);

    // edge_out = score @ Oe + edge_attr, with BN1(edge) stats
    k_gemm<64, 64, false, true, true, false, true, false>
        <<<gemmGrid(EG), 256, 0, stream>>>(ehin, OeH, nullptr, ea,
            ebuf, nullptr, stats + 128, stats + 192, (int)EG);

    // node_out = (wv/z) @ Oh + x, with BN1(node) stats
    k_nodenorm<<<grid1(N * D), 256, 0, stream>>>(wv, z, nh, N * D);
    k_gemm<64, 64, false, true, true, false, true, false>
        <<<gemmGrid(N), 256, 0, stream>>>(nh, OhH, nullptr, x,
            nbuf, nullptr, stats + 0, stats + 64, (int)N);

    // BN1
    k_bnfin<<<1, 64, 0, stream>>>(stats + 0,   stats + 64,  1.0f / (float)N);
    k_bnfin<<<1, 64, 0, stream>>>(stats + 128, stats + 192, 1.0f / (float)EG);
    k_bnapply<true><<<grid1(N * D), 256, 0, stream>>>(nbuf, stats + 0, stats + 64,
            nx_g, nx_b, nn16, N * D);
    k_bnapply<true><<<grid1(EG * D), 256, 0, stream>>>(ebuf, stats + 128, stats + 192,
            ne_g, ne_b, ehin, EG * D);

    // FFN node: relu(node @ W1 + b1) @ W2 + b2 + node, BN2(node) stats
    k_gemm<64, 128, true, false, false, true, false, true>
        <<<gemmGrid(N), 256, 0, stream>>>(nn16, fnW1H, fnb1, nullptr,
            nullptr, hidn, nullptr, nullptr, (int)N);
    k_gemm<128, 64, true, true, true, false, true, false>
        <<<gemmGrid(N), 256, 0, stream>>>(hidn, fnW2H, fnb2, nbuf,
            nodeOut, nullptr, stats + 256, stats + 320, (int)N);

    // FFN edge
    k_gemm<64, 128, true, false, false, true, false, true>
        <<<gemmGrid(EG), 256, 0, stream>>>(ehin, feW1H, feb1, nullptr,
            nullptr, hide, nullptr, nullptr, (int)EG);
    k_gemm<128, 64, true, true, true, false, true, false>
        <<<gemmGrid(EG), 256, 0, stream>>>(hide, feW2H, feb2, ebuf,
            edgeOut, nullptr, stats + 384, stats + 448, (int)EG);

    // BN2 (in place on d_out)
    k_bnfin<<<1, 64, 0, stream>>>(stats + 256, stats + 320, 1.0f / (float)N);
    k_bnfin<<<1, 64, 0, stream>>>(stats + 384, stats + 448, 1.0f / (float)EG);
    k_bnapply<false><<<grid1(N * D), 256, 0, stream>>>(nodeOut, stats + 256, stats + 320,
            fn_g, fn_b, nullptr, N * D);
    k_bnapply<false><<<grid1(EG * D), 256, 0, stream>>>(edgeOut, stats + 384, stats + 448,
            fe_g, fe_b, nullptr, EG * D);
}
